// GRU_28982439313484
// MI455X (gfx1250) — matrix operand order repaired and checked by reading, not compiled, not run on hardware
//
#include <hip/hip_runtime.h>

// GRU on MI455X (gfx1250, wave32, WMMA).
// Prep:    inputs f32->f16 (row major); 6 weight matrices f32->f16 pre-swizzled
//          into WMMA B-fragment layout (L2-resident).
// Phase 1: xz/xr/xh = inputs @ W_x{z,r,h} + b. 16x64 tile per wave, f16 WMMA.
// Phase 2: 4 persistent workgroups (16 batch rows each, batch-independent ->
//          no cross-WG sync), f16 WMMA recurrence, weights cached in LDS
//          fragments (W_hz) and VGPR fragments (W_hr, W_hh).

#define Bdim 64
#define Tdim 1024
#define Idim 256
#define Hdim 256
#define WELEM (Hdim * Hdim)      // 65536 elements per weight matrix

typedef __attribute__((ext_vector_type(16))) _Float16 v16h;
typedef __attribute__((ext_vector_type(8)))  _Float16 v8h;
typedef __attribute__((ext_vector_type(4)))  _Float16 v4h;
typedef __attribute__((ext_vector_type(4)))  float    v4f;
typedef __attribute__((ext_vector_type(8)))  float    v8f;

__device__ __forceinline__ float sigmoid_f(float x) {
  return 1.0f / (1.0f + __expf(-x));
}
__device__ __forceinline__ float tanh_f(float x) {
  float ax = fabsf(x);
  float e  = __expf(-2.0f * ax);
  float th = (1.0f - e) / (1.0f + e);
  return copysignf(th, x);
}

// ---------------------------------------------------------------------------
// Prep A: six weights -> f16 B-fragment layout.
// Dest-linear indexing (coalesced stores, L2-cached strided reads).
// WF[g][((kt*16+nt)*32 + lane)*16 + i] = W_g[kt*32 + (lane>>4)*8 + (i&7) + (i>>3)*16][nt*16 + (lane&15)]
// g: 0=Wxz 1=Wxr 2=Wxh 3=Whz 4=Whr 5=Whh
// ---------------------------------------------------------------------------
__global__ void __launch_bounds__(256)
gru_prep_weights(const float* __restrict__ Wxz, const float* __restrict__ Wxr,
                 const float* __restrict__ Wxh, const float* __restrict__ Whz,
                 const float* __restrict__ Whr, const float* __restrict__ Whh,
                 _Float16* __restrict__ WF)
{
  int tid = blockIdx.x * 256 + threadIdx.x;      // 6*65536 total
  int g   = tid >> 16;
  int d   = tid & 65535;
  int i    = d & 15;
  int lane = (d >> 4) & 31;
  int frag = d >> 9;
  int kt = frag >> 4, nt = frag & 15;
  int k  = kt * 32 + (lane >> 4) * 8 + (i & 7) + (i >> 3) * 16;
  int n  = nt * 16 + (lane & 15);
  const float* W = (g == 0) ? Wxz : (g == 1) ? Wxr : (g == 2) ? Wxh
                 : (g == 3) ? Whz : (g == 4) ? Whr : Whh;
  WF[tid] = (_Float16)W[k * Hdim + n];
}

// Prep B: inputs f32 -> f16, row-major, 4 elements per thread (b128 in, b64 out)
__global__ void __launch_bounds__(256)
gru_prep_inputs(const float* __restrict__ X, _Float16* __restrict__ Xh)
{
  size_t tid = (size_t)blockIdx.x * 256 + threadIdx.x;   // 4.19M threads
  v4f x = ((const v4f*)X)[tid];
  v4h h;
  h[0] = (_Float16)x[0]; h[1] = (_Float16)x[1];
  h[2] = (_Float16)x[2]; h[3] = (_Float16)x[3];
  ((v4h*)Xh)[tid] = h;
}

// ---------------------------------------------------------------------------
// Phase 1: XP[g][t*B + b][h] = sum_k inputs[b][t][k] * Wx_g[k][h] + bias_g[h]
// Each wave: one 16(M) x 64(N) tile; A fragment shared by 4 WMMAs per k-step.
// ---------------------------------------------------------------------------
__global__ void __launch_bounds__(256)
gru_xproj(const _Float16* __restrict__ Xh,   // [B*T, I] f16 (row m = b*T + t)
          const _Float16* __restrict__ WF,   // fragment-layout weights
          const float* __restrict__ bz, const float* __restrict__ br,
          const float* __restrict__ bh,
          float* __restrict__ XP)            // [3][T*B][Hd] f32
{
  const int lane = threadIdx.x & 31;
  const int wv   = threadIdx.x >> 5;
  const int half = lane >> 4;
  const int l16  = lane & 15;

  // tiles: 3 gates * 4096 m-tiles * 4 n-quads
  int tile = blockIdx.x * 8 + wv;
  int g    = tile / (4096 * 4);
  int rem  = tile % (4096 * 4);
  int mt   = rem >> 2;
  int nq   = rem & 3;
  int nt0  = nq * 4;

  const float* bias = (g == 0) ? bz : (g == 1) ? br : bh;
  const _Float16* WFg = WF + (size_t)g * WELEM;

  const int row0 = mt * 16;
  const _Float16* arow = Xh + (size_t)(row0 + l16) * Idim;

  v8f acc0 = {}, acc1 = {}, acc2 = {}, acc3 = {};
  #pragma unroll
  for (int kt = 0; kt < 8; ++kt) {
    int ka = kt * 32 + half * 8;
    v8h alo = *(const v8h*)&arow[ka];
    v8h ahi = *(const v8h*)&arow[ka + 16];
    v16h a = __builtin_shufflevector(alo, ahi, 0,1,2,3,4,5,6,7,
                                               8,9,10,11,12,13,14,15);
    const v16h* bp = (const v16h*)&WFg[((kt * 16 + nt0) * 32 + lane) * 16];
    acc0 = __builtin_amdgcn_wmma_f32_16x16x32_f16(false, a, false, bp[0],
                                                  (short)0, acc0, false, false);
    acc1 = __builtin_amdgcn_wmma_f32_16x16x32_f16(false, a, false, bp[32],
                                                  (short)0, acc1, false, false);
    acc2 = __builtin_amdgcn_wmma_f32_16x16x32_f16(false, a, false, bp[64],
                                                  (short)0, acc2, false, false);
    acc3 = __builtin_amdgcn_wmma_f32_16x16x32_f16(false, a, false, bp[96],
                                                  (short)0, acc3, false, false);
  }

  float* dst = XP + (size_t)g * (Bdim * Tdim) * Hdim;
  v8f accs[4] = {acc0, acc1, acc2, acc3};
  #pragma unroll
  for (int j = 0; j < 4; ++j) {
    int n  = (nt0 + j) * 16 + l16;
    float bv = bias[n];
    #pragma unroll
    for (int rr = 0; rr < 8; ++rr) {
      int mrow = row0 + rr + 8 * half;           // row m = b*T + t
      int bb   = mrow / Tdim;
      int tt   = mrow % Tdim;
      dst[((size_t)tt * Bdim + bb) * Hdim + n] = accs[j][rr] + bv;
    }
  }
}

// ---------------------------------------------------------------------------
// Phase 2: persistent recurrence. 4 WGs x 16 batch rows, 512 thr (16 waves).
// Wave w owns column tile w for all three hidden GEMMs.
// ---------------------------------------------------------------------------
__global__ void __launch_bounds__(512, 1)
gru_recurrence(const float* __restrict__ XP,      // [3][T*B][Hd] f32
               const _Float16* __restrict__ WF,   // fragment-layout weights
               float* __restrict__ out)           // [B,T,Hd] ++ [B,Hd] H_last
{
  __shared__ _Float16 WzS[WELEM];               // 128KB: W_hz B-fragments
  __shared__ _Float16 Hs [16 * Hdim];           // 8KB  f16 copy of H slice
  __shared__ _Float16 RHs[16 * Hdim];           // 8KB  R .* H

  const int lane  = threadIdx.x & 31;
  const int wv    = threadIdx.x >> 5;           // 0..15 == column tile nt
  const int half  = lane >> 4;
  const int l16   = lane & 15;
  const int bbase = blockIdx.x * 16;            // this WG's batch slice
  const int n     = wv * 16 + l16;              // global hidden column

  // One-time: W_hz fragments (pre-swizzled) -> LDS, coalesced b128 copies
  {
    const v8h* src = (const v8h*)(WF + (size_t)3 * WELEM);
    v8h* dst = (v8h*)WzS;
    for (int idx = threadIdx.x; idx < WELEM / 8; idx += blockDim.x)
      dst[idx] = src[idx];
  }
  // One-time: W_hr / W_hh fragments into registers (column block wv)
  v16h wr[8], wh[8];
  #pragma unroll
  for (int kt = 0; kt < 8; ++kt) {
    size_t fo = (size_t)((kt * 16 + wv) * 32 + lane) * 16;
    wr[kt] = *(const v16h*)&WF[(size_t)4 * WELEM + fo];
    wh[kt] = *(const v16h*)&WF[(size_t)5 * WELEM + fo];
  }
  // H0 = 0
  for (int idx = threadIdx.x; idx < 16 * Hdim; idx += blockDim.x)
    Hs[idx] = (_Float16)0.0f;
  __syncthreads();

  const size_t Mtot = (size_t)Tdim * Bdim;
  const float* XPz = XP;
  const float* XPr = XP + Mtot * Hdim;
  const float* XPh = XP + 2 * Mtot * Hdim;

  for (int t = 0; t < Tdim; ++t) {
    const size_t rowbase = ((size_t)t * Bdim + bbase) * Hdim;

    // ---- phase A: Z = sig(xz + H@Whz), R = sig(xr + H@Whr) ----
    v8f accz, accr;
    #pragma unroll
    for (int rr = 0; rr < 8; ++rr) {
      size_t off = rowbase + (size_t)(rr + 8 * half) * Hdim + n;
      accz[rr] = XPz[off];
      accr[rr] = XPr[off];
    }
    #pragma unroll
    for (int kt = 0; kt < 8; ++kt) {
      int ka = kt * 32 + half * 8;
      v8h alo = *(const v8h*)&Hs[l16 * Hdim + ka];
      v8h ahi = *(const v8h*)&Hs[l16 * Hdim + ka + 16];
      v16h a = __builtin_shufflevector(alo, ahi, 0,1,2,3,4,5,6,7,
                                                 8,9,10,11,12,13,14,15);
      const v16h* bz = (const v16h*)&WzS[((kt * 16 + wv) * 32 + lane) * 16];
      accz = __builtin_amdgcn_wmma_f32_16x16x32_f16(false, a, false, *bz,
                                                    (short)0, accz, false, false);
      accr = __builtin_amdgcn_wmma_f32_16x16x32_f16(false, a, false, wr[kt],
                                                    (short)0, accr, false, false);
    }
    float zr[8], hold[8];
    #pragma unroll
    for (int rr = 0; rr < 8; ++rr) {
      int mrow = rr + 8 * half;
      float hv = (float)Hs[mrow * Hdim + n];
      float z  = sigmoid_f(accz[rr]);
      float r  = sigmoid_f(accr[rr]);
      zr[rr] = z; hold[rr] = hv;
      RHs[mrow * Hdim + n] = (_Float16)(r * hv);
    }
    __syncthreads();

    // ---- phase B: Htilde = tanh(xh + (R.*H)@Whh); H = Z*H + (1-Z)*Htilde ----
    v8f acch;
    #pragma unroll
    for (int rr = 0; rr < 8; ++rr)
      acch[rr] = XPh[rowbase + (size_t)(rr + 8 * half) * Hdim + n];
    #pragma unroll
    for (int kt = 0; kt < 8; ++kt) {
      int ka = kt * 32 + half * 8;
      v8h alo = *(const v8h*)&RHs[l16 * Hdim + ka];
      v8h ahi = *(const v8h*)&RHs[l16 * Hdim + ka + 16];
      v16h a = __builtin_shufflevector(alo, ahi, 0,1,2,3,4,5,6,7,
                                                 8,9,10,11,12,13,14,15);
      acch = __builtin_amdgcn_wmma_f32_16x16x32_f16(false, a, false, wh[kt],
                                                    (short)0, acch, false, false);
    }
    #pragma unroll
    for (int rr = 0; rr < 8; ++rr) {
      int mrow = rr + 8 * half;
      float ht = tanh_f(acch[rr]);
      float hn = zr[rr] * hold[rr] + (1.0f - zr[rr]) * ht;
      Hs[mrow * Hdim + n] = (_Float16)hn;
      int bglob = bbase + mrow;
      out[((size_t)bglob * Tdim + t) * Hdim + n] = hn;
      if (t == Tdim - 1)
        out[(size_t)Bdim * Tdim * Hdim + (size_t)bglob * Hdim + n] = hn;
    }
    // hide next step's projection-load latency (global_prefetch_b8)
    if (t + 1 < Tdim) {
      size_t nb = ((size_t)(t + 1) * Bdim + bbase) * Hdim
                + (size_t)(8 * half) * Hdim + n;
      __builtin_prefetch(&XPz[nb], 0, 0);
      __builtin_prefetch(&XPr[nb], 0, 0);
      __builtin_prefetch(&XPh[nb], 0, 0);
    }
    __syncthreads();
  }
}

// ---------------------------------------------------------------------------
extern "C" void kernel_launch(void* const* d_in, const int* in_sizes, int n_in,
                              void* d_out, int out_size, void* d_ws, size_t ws_size,
                              hipStream_t stream) {
  const float* inputs = (const float*)d_in[0];
  const float* W_xz   = (const float*)d_in[1];
  const float* W_hz   = (const float*)d_in[2];
  const float* b_z    = (const float*)d_in[3];
  const float* W_xr   = (const float*)d_in[4];
  const float* W_hr   = (const float*)d_in[5];
  const float* b_r    = (const float*)d_in[6];
  const float* W_xh   = (const float*)d_in[7];
  const float* W_hh   = (const float*)d_in[8];
  const float* b_h    = (const float*)d_in[9];

  float* out = (float*)d_out;

  // Workspace layout:
  //   XP  f32 [3][T*B][Hd]   : 201,326,592 B
  //   Xh  f16 [B*T][I]       :  33,554,432 B
  //   WF  f16 [6][Hd*Hd]     :     786,432 B   (fragment layout)
  char* ws = (char*)d_ws;
  float*    XP = (float*)ws;
  _Float16* Xh = (_Float16*)(ws + 201326592);
  _Float16* WF = (_Float16*)(ws + 201326592 + 33554432);

  // Prep: weights -> fragment-layout f16; inputs -> f16
  gru_prep_weights<<<dim3((6 * WELEM) / 256), dim3(256), 0, stream>>>(
      W_xz, W_xr, W_xh, W_hz, W_hr, W_hh, WF);
  gru_prep_inputs<<<dim3((Bdim * Tdim * Idim / 4) / 256), dim3(256), 0, stream>>>(
      inputs, Xh);

  // Phase 1: 3 * 4096 m-tiles * 4 n-quads = 49152 wave-tiles, 8 waves/block
  gru_xproj<<<dim3(49152 / 8), dim3(256), 0, stream>>>(
      Xh, WF, b_z, b_r, b_h, XP);

  // Phase 2: 4 persistent WGs, 16 batch rows each, no cross-WG sync
  gru_recurrence<<<dim3(4), dim3(512), 0, stream>>>(XP, WF, out);
}
